// GcnSAGELayer_3444563771449
// MI455X (gfx1250) — matrix operand order repaired and checked by reading, not compiled, not run on hardware
//
#include <hip/hip_runtime.h>
#include <hip/hip_bf16.h>

// GCN-SAGE layer for MI455X (gfx1250, wave32, WMMA).
// Pipeline: zero ws -> Wt f16 transpose (once, L2-resident)
//        -> edge scatter (1 wave per edge, f32 atomics)
//        -> fused [h|r|bt|l|r] @ W^T + b -> LayerNorm -> ReLU using
//           v_wmma_f32_16x16x32_f16 (f16 inputs, f32 accumulate over K=640),
//           A tile double-buffered in LDS, B fragments straight from global.

#define N_NODES 50000
#define N_EDGES 800000
#define DFEAT   128
#define K_TOT   640
#define LN_EPS  1e-5f

typedef _Float16 v16h __attribute__((ext_vector_type(16)));
typedef _Float16 v8h  __attribute__((ext_vector_type(8)));
typedef _Float16 v4h  __attribute__((ext_vector_type(4)));
typedef float    v8f  __attribute__((ext_vector_type(8)));

// ---------------------------------------------------------------- zero ws
__global__ void zero_ws_kernel(float4* __restrict__ ws, int n4) {
  int i = blockIdx.x * blockDim.x + threadIdx.x;
  if (i < n4) ws[i] = make_float4(0.f, 0.f, 0.f, 0.f);
}

// ------------------------------------------------- W^T -> f16, k-major
// Wt[k][n] = (f16) W[n][k].  640x128 f16 = 160 KB: lives in L2 for the
// whole GEMM, so B fragments can be fetched directly from global.
__global__ void wt_f16_kernel(const float* __restrict__ W,
                              _Float16* __restrict__ Wt) {
  int e = blockIdx.x * blockDim.x + threadIdx.x;
  if (e < K_TOT * DFEAT) {
    int k = e >> 7, n = e & 127;
    Wt[e] = (_Float16)W[n * K_TOT + k];
  }
}

// ------------------------------------------------------------ edge scatter
// One wave (32 lanes) per edge: lanes cooperatively move one 128-float row
// (512 B contiguous) and atomically add the dist-scaled row into the
// direction buffer selected by positions[e].
__global__ __launch_bounds__(256)
void edge_scatter_kernel(const float* __restrict__ h,
                         const float* __restrict__ dist,
                         const int*   __restrict__ src,
                         const int*   __restrict__ dst,
                         const int*   __restrict__ pos,
                         float* __restrict__ aggR,
                         float* __restrict__ aggB,
                         float* __restrict__ aggL) {
  int e    = (blockIdx.x * blockDim.x + threadIdx.x) >> 5;
  int lane = threadIdx.x & 31;
  if (e >= N_EDGES) return;
  int p = pos[e];
  if (p >= 3) return;  // position 3 contributes to no direction
  float w = dist[e];
  const float4* hv = (const float4*)(h + (size_t)src[e] * DFEAT);
  float* base = (p == 0) ? aggR : (p == 1) ? aggB : aggL;
  float4 v = hv[lane];
  float* o = base + (size_t)dst[e] * DFEAT + lane * 4;
  atomicAdd(o + 0, v.x * w);
  atomicAdd(o + 1, v.y * w);
  atomicAdd(o + 2, v.z * w);
  atomicAdd(o + 3, v.w * w);
}

// --------------------------------------------- fused GEMM + LayerNorm + ReLU
// Block = 256 threads (8 waves) -> one 64x128 output tile (64 nodes).
// Wave w: rows rb*16..+15 (rb=w&3), cols cbase..cbase+63 (cbase=(w>>2)*64),
// 4 x v8f accumulators.  Outer loop over the 5 cat segments (pointer picked
// once), inner 4 K-steps of 32.  A tile (64x32 f16, 4 KB) double-buffered in
// LDS -> one barrier per K-step; B fragments loaded from global Wt.
__global__ __launch_bounds__(256)
void gemm_ln_kernel(const float* __restrict__ h,
                    const _Float16* __restrict__ Wt,   // [640][128] f16
                    const float* __restrict__ bias,
                    const float* __restrict__ gamma,
                    const float* __restrict__ beta,
                    const float* __restrict__ aggR,
                    const float* __restrict__ aggB,
                    const float* __restrict__ aggL,
                    float* __restrict__ out) {
  __shared__ float smem[64 * 128];                 // 32 KB, dual-purpose
  _Float16* const Ab0 = (_Float16*)smem;           // A buffer 0: [64][32]
  _Float16* const Ab1 = (_Float16*)smem + 64 * 32; // A buffer 1: [64][32]

  const int tid      = threadIdx.x;
  const int lane     = tid & 31;
  const int wave     = tid >> 5;
  const int nodeBase = blockIdx.x * 64;

  const int rb    = wave & 3;          // 16-row block within the 64-row tile
  const int cbase = (wave >> 2) * 64;  // 64-col half

  // Per-lane A fragment coords (ISA 7.12.2, 16-bit A 16x32):
  //   lane l: M = l%16; K runs = [K0, K0+7] and [K0+16, K0+23], K0 = (l<16?0:8)
  const int am  = rb * 16 + (lane & 15);
  const int ak0 = (lane & 16) ? 8 : 0;

  // Per-thread A staging coords: 2 float4 slots of the 64x32 tile.
  const int s0 = tid * 2;
  const int sm = s0 >> 3;          // row (same for both slots: 8 slots/row)
  const int sk = (s0 & 7) * 4;     // k offset of first float4

  v8f acc[4] = {};

  for (int seg = 0; seg < 5; ++seg) {
    const float* sp = (seg == 0) ? h
                    : (seg == 1) ? aggR
                    : (seg == 2) ? aggB
                    : (seg == 3) ? aggL
                                 : aggR;  // "top" faithfully reuses r
    const float* srow = (nodeBase + sm < N_NODES)
                      ? sp + (size_t)(nodeBase + sm) * DFEAT : nullptr;

    for (int k4 = 0; k4 < 4; ++k4) {
      const int kt = seg * 4 + k4;
      _Float16* const As = (kt & 1) ? Ab1 : Ab0;

      // Stage A: two float4 loads -> f16 -> LDS (writers target the buffer
      // NOT being read this step, so only one barrier per step is needed).
#pragma unroll
      for (int i = 0; i < 2; ++i) {
        float4 v = make_float4(0.f, 0.f, 0.f, 0.f);
        if (srow) v = *(const float4*)(srow + k4 * 32 + sk + i * 4);
        v4h hv = {(_Float16)v.x, (_Float16)v.y, (_Float16)v.z, (_Float16)v.w};
        *(v4h*)(As + sm * 32 + sk + i * 4) = hv;
      }
      __syncthreads();

      // A fragment (LDS, DScnt) ...
      v8h alo = *(const v8h*)(As + am * 32 + ak0);
      v8h ahi = *(const v8h*)(As + am * 32 + ak0 + 16);
      v16h afrag = __builtin_shufflevector(alo, ahi, 0, 1, 2, 3, 4, 5, 6, 7,
                                           8, 9, 10, 11, 12, 13, 14, 15);

      // ... B fragments (global Wt, LOADcnt; lane%16 = N, element = K in A's order).
      const _Float16* bcol = Wt + (size_t)(kt * 32 + ak0) * DFEAT + cbase + (lane & 15);
      v16h bfrag[4];
#pragma unroll
      for (int ct = 0; ct < 4; ++ct) {
#pragma unroll
        for (int s = 0; s < 8; ++s) {
          bfrag[ct][s]     = bcol[s * DFEAT + ct * 16];
          bfrag[ct][8 + s] = bcol[(16 + s) * DFEAT + ct * 16];
        }
      }
#pragma unroll
      for (int ct = 0; ct < 4; ++ct)
        acc[ct] = __builtin_amdgcn_wmma_f32_16x16x32_f16(
            false, afrag, false, bfrag[ct], (short)0, acc[ct], false, false);
    }
  }

  // Spill z tile to LDS per the 16x16 f32 C/D layout:
  // VGPR r, lane l: M = r + (l<16 ? 0 : 8), N = l%16.
  __syncthreads();
  float* zs = smem;
#pragma unroll
  for (int ct = 0; ct < 4; ++ct) {
    int col    = cbase + ct * 16 + (lane & 15);
    int rowoff = rb * 16 + ((lane >> 4) << 3);
#pragma unroll
    for (int r = 0; r < 8; ++r)
      zs[(rowoff + r) * 128 + col] = acc[ct][r];
  }
  __syncthreads();

  // LayerNorm + affine + ReLU: 4 lanes per row (256 threads / 64 rows).
  // xor-masks 1 and 2 stay inside each 4-lane group on wave32.
  int row = tid >> 2;
  int sub = tid & 3;
  float vbuf[32];
  float sum = 0.f, sumsq = 0.f;
#pragma unroll
  for (int j = 0; j < 32; ++j) {
    int col = sub * 32 + j;
    float v = zs[row * 128 + col] + bias[col];
    vbuf[j] = v;
    sum += v;
    sumsq += v * v;
  }
  sum   += __shfl_xor(sum, 1);   sum   += __shfl_xor(sum, 2);
  sumsq += __shfl_xor(sumsq, 1); sumsq += __shfl_xor(sumsq, 2);
  float mean = sum * (1.f / 128.f);
  float var  = sumsq * (1.f / 128.f) - mean * mean;
  float rstd = rsqrtf(var + LN_EPS);
  int node = nodeBase + row;
  if (node < N_NODES) {
#pragma unroll
    for (int j = 0; j < 32; ++j) {
      int col = sub * 32 + j;
      float v = (vbuf[j] - mean) * rstd * gamma[col] + beta[col];
      out[(size_t)node * 128 + col] = v > 0.f ? v : 0.f;
    }
  }
}

// ------------------------------------------------------------------ launch
extern "C" void kernel_launch(void* const* d_in, const int* in_sizes, int n_in,
                              void* d_out, int out_size, void* d_ws, size_t ws_size,
                              hipStream_t stream) {
  const float* h     = (const float*)d_in[0];
  const float* dist  = (const float*)d_in[1];
  const float* W     = (const float*)d_in[2];
  const float* b     = (const float*)d_in[3];
  const float* gamma = (const float*)d_in[4];
  const float* beta  = (const float*)d_in[5];
  const int*   src   = (const int*)d_in[6];
  const int*   dst   = (const int*)d_in[7];
  const int*   pos   = (const int*)d_in[8];

  // Workspace: 3 x [N,128] f32 direction accumulators (76.8 MB) + Wt f16.
  float* aggR = (float*)d_ws;
  float* aggB = aggR + (size_t)N_NODES * DFEAT;
  float* aggL = aggB + (size_t)N_NODES * DFEAT;
  _Float16* Wt = (_Float16*)(aggL + (size_t)N_NODES * DFEAT);

  int n4 = 3 * N_NODES * DFEAT / 4;
  zero_ws_kernel<<<(n4 + 255) / 256, 256, 0, stream>>>((float4*)d_ws, n4);

  wt_f16_kernel<<<(K_TOT * DFEAT + 255) / 256, 256, 0, stream>>>(W, Wt);

  int edge_threads = N_EDGES * 32;
  edge_scatter_kernel<<<(edge_threads + 255) / 256, 256, 0, stream>>>(
      h, dist, src, dst, pos, aggR, aggB, aggL);

  gemm_ln_kernel<<<(N_NODES + 63) / 64, 256, 0, stream>>>(
      h, Wt, b, gamma, beta, aggR, aggB, aggL, (float*)d_out);
}
